// ImgModel_65369402245605
// MI455X (gfx1250) — matrix operand order repaired and checked by reading, not compiled, not run on hardware
//
#include <hip/hip_runtime.h>
#include <hip/hip_bf16.h>

// ---------------------------------------------------------------------------
// MI455X / gfx1250 implementation.
// Heavy compute via V_WMMA_F32_16X16X32_BF16 with 64x64 register blocking per
// wave (4x4 fragments, 16 accumulators): 32 flops/byte of operand traffic,
// 4x the arithmetic intensity of a single-fragment loop.  The spatial
// 3200x3200 f32 affinity matrix (41MB) is processed one batch at a time so it
// stays resident in the 192MB L2; softmax statistics and both sigmoid gates
// are algebraically fused into online reduction passes over that matrix.
// ---------------------------------------------------------------------------

#define BS_  8
#define CH_  256
#define NSP_ 3200   // 40*80

typedef __attribute__((ext_vector_type(16))) __bf16        v16bf;
typedef __attribute__((ext_vector_type(8)))  float         v8f;
typedef __attribute__((ext_vector_type(4)))  unsigned int  v4u;

union BF16Frag { v16bf v; v4u u[2]; };

__device__ __forceinline__ unsigned short f2bf(float f) {
    unsigned int u = __float_as_uint(f);
    unsigned int r = u + 0x7FFFu + ((u >> 16) & 1u);   // round-to-nearest-even
    return (unsigned short)(r >> 16);
}
__device__ __forceinline__ float bf2f(unsigned short h) {
    return __uint_as_float(((unsigned int)h) << 16);
}
__device__ __forceinline__ float wave_max(float v) {
    #pragma unroll
    for (int m = 16; m > 0; m >>= 1) v = fmaxf(v, __shfl_xor(v, m, 32));
    return v;
}
__device__ __forceinline__ float wave_sum(float v) {
    #pragma unroll
    for (int m = 16; m > 0; m >>= 1) v += __shfl_xor(v, m, 32);
    return v;
}

// ---------------------------------------------------------------------------
// Wave-level 64x64 bf16 GEMM mainloop: acc[i][j] += A(64xK) * B(64xK)^T
// A: M-rows K-contiguous (lda elems/row).  B: N-rows K-contiguous.
// 16-bit A-fragment lane layout per CDNA5 ISA (16x32): lane l16
// = row, half 0/1 selects K {0..7,16..23} / {8..15,24..31}.  B mirrored with
// the same K split: lanes0-15 K {0..7,16..23}, lanes16-31 K {8..15,24..31}.
// ---------------------------------------------------------------------------
__device__ __forceinline__ void wmma_block64(const unsigned short* __restrict__ A, int lda,
                                             const unsigned short* __restrict__ B, int ldb,
                                             int m0, int n0, int K, v8f (&acc)[4][4]) {
    const int lane = threadIdx.x & 31;
    const int half = lane >> 4;      // 0 | 1
    const int l16  = lane & 15;
    const unsigned short* arow[4];
    const unsigned short* brow[4];
    #pragma unroll
    for (int i = 0; i < 4; ++i) {
        arow[i] = A + (size_t)(m0 + 16 * i + l16) * lda;
        brow[i] = B + (size_t)(n0 + 16 * i + l16) * ldb;
    }
    const int aoff = half * 8;       // A: K = {0..7 | 8..15} then +16
    const int boff = half * 8;       // B: K = {0..7 | 8..15} then +16
    for (int k = 0; k < K; k += 32) {
        BF16Frag a[4];
        #pragma unroll
        for (int i = 0; i < 4; ++i) {
            a[i].u[0] = *(const v4u*)(arow[i] + k + aoff);
            a[i].u[1] = *(const v4u*)(arow[i] + k + 16 + aoff);
        }
        #pragma unroll
        for (int j = 0; j < 4; ++j) {
            BF16Frag b;
            b.u[0] = *(const v4u*)(brow[j] + k + boff);
            b.u[1] = *(const v4u*)(brow[j] + k + 16 + boff);
            #pragma unroll
            for (int i = 0; i < 4; ++i) {
                acc[i][j] = __builtin_amdgcn_wmma_f32_16x16x32_bf16(
                    false, a[i].v, false, b.v, (short)0, acc[i][j], false, false);
            }
        }
    }
}

// C/D layout: VGPR r, lanes0-15 -> (M=r, N=l16); lanes16-31 -> (M=r+8, N=l16)

__global__ __launch_bounds__(32)
void gemm64_f32out(const unsigned short* __restrict__ A, int lda, size_t sA,
                   const unsigned short* __restrict__ B, int ldb, size_t sB,
                   float* __restrict__ C, int ldc, size_t sC,
                   int nt64, int K, const float* __restrict__ scale, size_t sScale) {
    const int tile = blockIdx.x;
    const int mt = tile / nt64, nt = tile % nt64;
    const int b = blockIdx.y;
    A += (size_t)b * sA; B += (size_t)b * sB; C += (size_t)b * sC;
    const int m0 = mt * 64, n0 = nt * 64;
    v8f acc[4][4] = {};
    wmma_block64(A, lda, B, ldb, m0, n0, K, acc);
    const int lane = threadIdx.x & 31;
    const int half = lane >> 4, l16 = lane & 15;
    #pragma unroll
    for (int j = 0; j < 4; ++j) {
        const int col = n0 + 16 * j + l16;
        float g = 1.0f;
        if (scale) g = scale[sScale * b + col];
        #pragma unroll
        for (int i = 0; i < 4; ++i) {
            #pragma unroll
            for (int r = 0; r < 8; ++r) {
                int row = m0 + 16 * i + r + half * 8;
                C[(size_t)row * ldc + col] = acc[i][j][r] * g;
            }
        }
    }
}

__global__ __launch_bounds__(32)
void gemm64_dualout(const unsigned short* __restrict__ A, int lda, size_t sA,
                    const unsigned short* __restrict__ B, int ldb, size_t sB,
                    unsigned short* __restrict__ Crow, int ldr,  size_t sCr, // [M][N] bf16
                    unsigned short* __restrict__ Ccol, int ldcc, size_t sCc, // [N][M] bf16
                    int nt64, int K, const float* __restrict__ scale, size_t sScale) {
    const int tile = blockIdx.x;
    const int mt = tile / nt64, nt = tile % nt64;
    const int b = blockIdx.y;
    A += (size_t)b * sA; B += (size_t)b * sB;
    Crow += (size_t)b * sCr; Ccol += (size_t)b * sCc;
    const int m0 = mt * 64, n0 = nt * 64;
    v8f acc[4][4] = {};
    wmma_block64(A, lda, B, ldb, m0, n0, K, acc);
    const int lane = threadIdx.x & 31;
    const int half = lane >> 4, l16 = lane & 15;
    #pragma unroll
    for (int j = 0; j < 4; ++j) {
        const int col = n0 + 16 * j + l16;
        const float g = scale[sScale * b + col];
        #pragma unroll
        for (int i = 0; i < 4; ++i) {
            #pragma unroll
            for (int r = 0; r < 8; ++r) {
                int row = m0 + 16 * i + r + half * 8;
                unsigned short h = f2bf(acc[i][j][r] * g);
                Crow[(size_t)row * ldr  + col] = h;
                Ccol[(size_t)col * ldcc + row] = h;
            }
        }
    }
}

// --------------------------- elementwise / softmax --------------------------

__global__ __launch_bounds__(256)
void convert_inputs(const float* __restrict__ er, const float* __restrict__ cb,
                    unsigned short* __restrict__ erbf, unsigned short* __restrict__ erT,
                    unsigned short* __restrict__ cbbf, unsigned short* __restrict__ cbT,
                    int total) {
    int idx = blockIdx.x * blockDim.x + threadIdx.x;
    if (idx >= total) return;
    int n = idx % NSP_;
    int t = idx / NSP_;
    int c = t % CH_;
    int b = t / CH_;
    size_t tp = ((size_t)b * NSP_ + n) * CH_ + c;
    unsigned short he = f2bf(er[idx]);
    unsigned short hc = f2bf(cb[idx]);
    erbf[idx] = he; erT[tp] = he;
    cbbf[idx] = hc; cbT[tp] = hc;
}

// a2 = softmax(ch_aff, axis=d) row-wise; also store row stats for the col pass
__global__ __launch_bounds__(32)
void row_softmax_256(const float* __restrict__ aff, unsigned short* __restrict__ outbf,
                     float* __restrict__ rmax, float* __restrict__ rsum) {
    const int row = blockIdx.x;                 // b*256 + c
    const float* p = aff + (size_t)row * 256;
    const int lane = threadIdx.x;
    float x[8], mx = -3.4e38f;
    #pragma unroll
    for (int i = 0; i < 8; ++i) { x[i] = p[lane + 32 * i]; mx = fmaxf(mx, x[i]); }
    mx = wave_max(mx);
    float s = 0.f;
    #pragma unroll
    for (int i = 0; i < 8; ++i) { x[i] = __expf(x[i] - mx); s += x[i]; }
    s = wave_sum(s);
    const float inv = 1.0f / s;
    #pragma unroll
    for (int i = 0; i < 8; ++i) outbf[(size_t)row * 256 + lane + 32 * i] = f2bf(x[i] * inv);
    if (lane == 0) { rmax[row] = mx; rsum[row] = s; }
}

// a1 = softmax(ch_aff, axis=c) column-wise; fused u1 = w_ch_er^T a1,
// u2 = w_ch_cb^T a2 (uses row stats).
__global__ __launch_bounds__(32)
void col_softmax_256(const float* __restrict__ aff, unsigned short* __restrict__ a1bf,
                     const float* __restrict__ rmax, const float* __restrict__ rsum,
                     const float* __restrict__ w1, const float* __restrict__ w2,
                     float* __restrict__ u1, float* __restrict__ u2) {
    const int gid = blockIdx.x;                 // b*256 + d
    const int b = gid >> 8, d = gid & 255;
    const float* base = aff + ((size_t)b << 16) + d;
    const int lane = threadIdx.x;
    float x[8], mx = -3.4e38f;
    #pragma unroll
    for (int i = 0; i < 8; ++i) { x[i] = base[(size_t)(lane + 32 * i) * 256]; mx = fmaxf(mx, x[i]); }
    mx = wave_max(mx);
    float s = 0.f, t2 = 0.f;
    #pragma unroll
    for (int i = 0; i < 8; ++i) {
        int c = lane + 32 * i, rc = (b << 8) + c;
        float xi = x[i];
        float e = __expf(xi - mx);
        s += e; x[i] = e;
        t2 += w2[c] * __expf(xi - rmax[rc]) / rsum[rc];
    }
    s = wave_sum(s); t2 = wave_sum(t2);
    const float inv = 1.0f / s;
    float t1 = 0.f;
    #pragma unroll
    for (int i = 0; i < 8; ++i) {
        int c = lane + 32 * i;
        float a = x[i] * inv;
        a1bf[(((size_t)b << 8) + c) * 256 + d] = f2bf(a);
        t1 += w1[c] * a;
    }
    t1 = wave_sum(t1);
    if (lane == 0) { u1[gid] = t1; u2[gid] = t2; }
}

// out[b,n] = mode ? sigmoid(dot(u_b, T[b,n,:]) + bias) : dot(u_b, T[b,n,:])
__global__ __launch_bounds__(256)
void gate_vec(const unsigned short* __restrict__ Tbf, const float* __restrict__ u,
              size_t uStride, const float* __restrict__ bias,
              float* __restrict__ out, int mode) {
    int idx = blockIdx.x * blockDim.x + threadIdx.x;
    if (idx >= BS_ * NSP_) return;
    int b = idx / NSP_;
    const unsigned short* t = Tbf + (size_t)idx * CH_;
    const float* uu = u + uStride * b;
    float acc = 0.f;
    for (int c = 0; c < CH_; ++c) acc += uu[c] * bf2f(t[c]);
    if (mode) acc = 1.0f / (1.0f + __expf(-(acc + bias[0])));
    out[idx] = acc;
}

// row stats over m for s2 (softmax axis=2)
__global__ __launch_bounds__(32)
void row_stats_N(const float* __restrict__ aff, float* __restrict__ rmax, float* __restrict__ rsum) {
    const int n = blockIdx.x;
    const float* p = aff + (size_t)n * NSP_;
    const int lane = threadIdx.x;
    float mx = -3.4e38f;
    for (int m = lane; m < NSP_; m += 32) mx = fmaxf(mx, p[m]);
    mx = wave_max(mx);
    float s = 0.f;
    for (int m = lane; m < NSP_; m += 32) s += __expf(p[m] - mx);
    s = wave_sum(s);
    if (lane == 0) { rmax[n] = mx; rsum[n] = s; }
}

// column pass over n for each m: online (max,sum,v1-weighted-sum) for s1 plus
// v2-weighted s2 sum -> both spatial gate vectors produced here.
__global__ __launch_bounds__(32)
void col_pass_N(const float* __restrict__ aff,
                const float* __restrict__ rmax2, const float* __restrict__ rsum2,
                const float* __restrict__ v1, const float* __restrict__ v2,
                const float* __restrict__ b1, const float* __restrict__ b2,
                float* __restrict__ cmax, float* __restrict__ csum,
                float* __restrict__ g1, float* __restrict__ g2) {
    const int m = blockIdx.x;
    const int lane = threadIdx.x;
    float M = -3.4e38f, S = 0.f, W1 = 0.f, W2 = 0.f;
    for (int n = lane; n < NSP_; n += 32) {
        float x = aff[(size_t)n * NSP_ + m];
        float nm = fmaxf(M, x);
        float sc = __expf(M - nm);
        float e  = __expf(x - nm);
        S  = S * sc + e;
        W1 = W1 * sc + v1[n] * e;
        M  = nm;
        W2 += v2[n] * __expf(x - rmax2[n]) / rsum2[n];
    }
    #pragma unroll
    for (int msk = 16; msk > 0; msk >>= 1) {
        float oM = __shfl_xor(M, msk, 32), oS = __shfl_xor(S, msk, 32), oW = __shfl_xor(W1, msk, 32);
        float nm = fmaxf(M, oM);
        float e1 = __expf(M - nm), e2 = __expf(oM - nm);
        S = S * e1 + oS * e2;
        W1 = W1 * e1 + oW * e2;
        M = nm;
    }
    W2 = wave_sum(W2);
    if (lane == 0) {
        cmax[m] = M; csum[m] = S;
        g1[m] = 1.0f / (1.0f + __expf(-(W1 / S + b1[0])));
        g2[m] = 1.0f / (1.0f + __expf(-(W2 + b2[0])));
    }
}

// s1T[m][n] = bf16(s1[n,m]), s2T[m][n] = bf16(s2[n,m]) — K-contiguous B operands
__global__ __launch_bounds__(256)
void exp_write(const float* __restrict__ aff,
               const float* __restrict__ cmax, const float* __restrict__ csum,
               const float* __restrict__ rmax, const float* __restrict__ rsum,
               unsigned short* __restrict__ s1T, unsigned short* __restrict__ s2T) {
    int idx = blockIdx.x * blockDim.x + threadIdx.x;   // = m*NSP_ + n
    if (idx >= NSP_ * NSP_) return;
    int n = idx % NSP_, m = idx / NSP_;
    float x = aff[(size_t)n * NSP_ + m];
    s1T[idx] = f2bf(__expf(x - cmax[m]) / csum[m]);
    s2T[idx] = f2bf(__expf(x - rmax[n]) / rsum[n]);
}

// ---------------------------------------------------------------------------

extern "C" void kernel_launch(void* const* d_in, const int* in_sizes, int n_in,
                              void* d_out, int out_size, void* d_ws, size_t ws_size,
                              hipStream_t stream) {
    (void)in_sizes; (void)n_in; (void)out_size; (void)ws_size;
    const int B = BS_, C = CH_, N = NSP_;

    const float* feat_er = (const float*)d_in[0];
    const float* feat_cb = (const float*)d_in[1];
    const float* w_ch_er = (const float*)d_in[2];
    const float* b_ch_er = (const float*)d_in[3];
    const float* w_ch_cb = (const float*)d_in[4];
    const float* b_ch_cb = (const float*)d_in[5];
    const float* w_sp_er = (const float*)d_in[6];
    const float* b_sp_er = (const float*)d_in[7];
    const float* w_sp_cb = (const float*)d_in[8];
    const float* b_sp_cb = (const float*)d_in[9];
    float* out = (float*)d_out;

    // ---- workspace carve-up (~190 MB) ----
    char* wsb = (char*)d_ws; size_t off = 0;
    auto alloc = [&](size_t bytes) -> void* {
        void* p = wsb + off;
        off += (bytes + 255) & ~(size_t)255;
        return p;
    };
    const size_t CN = (size_t)C * N, CC = (size_t)C * C;
    unsigned short* erbf   = (unsigned short*)alloc(B * CN * 2); // [b][c][n] bf16
    unsigned short* erTbf  = (unsigned short*)alloc(B * CN * 2); // [b][n][c]
    unsigned short* cbbf   = (unsigned short*)alloc(B * CN * 2);
    unsigned short* cbTbf  = (unsigned short*)alloc(B * CN * 2);
    float*          chaff  = (float*)alloc(B * CC * 4);
    unsigned short* a2bf   = (unsigned short*)alloc(B * CC * 2);
    unsigned short* a1bf   = (unsigned short*)alloc(B * CC * 2);
    float* rmax_ch = (float*)alloc(B * C * 4);
    float* rsum_ch = (float*)alloc(B * C * 4);
    float* u1      = (float*)alloc(B * C * 4);
    float* u2      = (float*)alloc(B * C * 4);
    float* gch_er  = (float*)alloc((size_t)B * N * 4);
    float* gch_cb  = (float*)alloc((size_t)B * N * 4);
    unsigned short* erf_bf  = (unsigned short*)alloc(B * CN * 2); // gated ch_er [b][c][n]
    unsigned short* erfT_bf = (unsigned short*)alloc(B * CN * 2); // [b][n][c]
    unsigned short* cbf_bf  = (unsigned short*)alloc(B * CN * 2);
    unsigned short* cbfT_bf = (unsigned short*)alloc(B * CN * 2);
    float* v1 = (float*)alloc((size_t)B * N * 4);
    float* v2 = (float*)alloc((size_t)B * N * 4);
    float* aff = (float*)alloc((size_t)N * N * 4);               // 41 MB, L2-resident
    float* rmax2 = (float*)alloc(N * 4);
    float* rsum2 = (float*)alloc(N * 4);
    float* cmax  = (float*)alloc(N * 4);
    float* csum  = (float*)alloc(N * 4);
    float* g1    = (float*)alloc(N * 4);
    float* g2    = (float*)alloc(N * 4);
    unsigned short* s1T = (unsigned short*)alloc((size_t)N * N * 2);
    unsigned short* s2T = (unsigned short*)alloc((size_t)N * N * 2);

    // ---- stage 0: bf16 operand copies (both orientations) ----
    int total = B * C * N;
    convert_inputs<<<(total + 255) / 256, 256, 0, stream>>>(
        feat_er, feat_cb, erbf, erTbf, cbbf, cbTbf, total);

    // ---- stage A: ch_aff = er @ cb^T  (M=N=256, K=3200), all batches ----
    gemm64_f32out<<<dim3(4 * 4, B), 32, 0, stream>>>(
        erbf, N, CN, cbbf, N, CN, chaff, C, CC, /*nt64*/4, /*K*/N, nullptr, 0);

    // ---- stage B: channel softmaxes + fused gate pre-vectors ----
    row_softmax_256<<<B * C, 32, 0, stream>>>(chaff, a2bf, rmax_ch, rsum_ch);
    col_softmax_256<<<B * C, 32, 0, stream>>>(chaff, a1bf, rmax_ch, rsum_ch,
                                              w_ch_er, w_ch_cb, u1, u2);
    gate_vec<<<(B * N + 255) / 256, 256, 0, stream>>>(erTbf, u1, C, b_ch_er, gch_er, 1);
    gate_vec<<<(B * N + 255) / 256, 256, 0, stream>>>(cbTbf, u2, C, b_ch_cb, gch_cb, 1);

    // ---- stage C: er_f = gate(a1 @ er), cb_f = gate(a2 @ cb); dual bf16 out ----
    gemm64_dualout<<<dim3(4 * 50, B), 32, 0, stream>>>(
        a1bf, C, CC, erTbf, C, CN, erf_bf, N, CN, erfT_bf, C, CN,
        /*nt64*/50, /*K*/C, gch_er, (size_t)N);
    gemm64_dualout<<<dim3(4 * 50, B), 32, 0, stream>>>(
        a2bf, C, CC, cbTbf, C, CN, cbf_bf, N, CN, cbfT_bf, C, CN,
        /*nt64*/50, /*K*/C, gch_cb, (size_t)N);

    // ---- spatial-gate pre-vectors: v1 = w_sp_er^T er_f, v2 = w_sp_cb^T cb_f ----
    gate_vec<<<(B * N + 255) / 256, 256, 0, stream>>>(erfT_bf, w_sp_er, 0, nullptr, v1, 0);
    gate_vec<<<(B * N + 255) / 256, 256, 0, stream>>>(cbfT_bf, w_sp_cb, 0, nullptr, v2, 0);

    // ---- spatial stage: one batch at a time so aff (41MB) stays in L2 ----
    for (int b = 0; b < B; ++b) {
        // sp_aff[n,m] = sum_c er_f[c,n] cb_f[c,m]   (M=N=3200, K=256)
        gemm64_f32out<<<dim3(50 * 50, 1), 32, 0, stream>>>(
            erfT_bf + (size_t)b * CN, C, 0, cbfT_bf + (size_t)b * CN, C, 0,
            aff, N, 0, /*nt64*/50, /*K*/C, nullptr, 0);

        row_stats_N<<<N, 32, 0, stream>>>(aff, rmax2, rsum2);
        col_pass_N<<<N, 32, 0, stream>>>(aff, rmax2, rsum2,
                                         v1 + (size_t)b * N, v2 + (size_t)b * N,
                                         b_sp_er, b_sp_cb, cmax, csum, g1, g2);
        exp_write<<<((size_t)N * N + 255) / 256, 256, 0, stream>>>(
            aff, cmax, csum, rmax2, rsum2, s1T, s2T);

        // sp_er = g1 ⊙ (er_f @ s1):  M=256, N=3200, K=3200
        gemm64_f32out<<<dim3(4 * 50, 1), 32, 0, stream>>>(
            erf_bf + (size_t)b * CN, N, 0, s1T, N, 0,
            out + (size_t)b * CN, N, 0, /*nt64*/50, /*K*/N, g1, 0);
        // sp_cb = g2 ⊙ (cb_f @ s2)
        gemm64_f32out<<<dim3(4 * 50, 1), 32, 0, stream>>>(
            cbf_bf + (size_t)b * CN, N, 0, s2T, N, 0,
            out + (size_t)(B + b) * CN, N, 0, /*nt64*/50, /*K*/N, g2, 0);
    }
}
